// Encoder_63402307223929
// MI455X (gfx1250) — compile-verified, matrix operand order checked by reading, not run on hardware
//
#include <hip/hip_runtime.h>

#define B_   64
#define T_   128
#define E_   1024
#define H_   1280
#define G3H  3840
#define NBLK2 20

typedef __attribute__((ext_vector_type(16))) __bf16 v16bf;
typedef __attribute__((ext_vector_type(8)))  float  v8f;
typedef __attribute__((ext_vector_type(4)))  unsigned int u32x4;
typedef __attribute__((ext_vector_type(8)))  int i32x8;
typedef __attribute__((ext_vector_type(4)))  int i32x4;

#if __has_builtin(__builtin_amdgcn_tensor_load_to_lds) && __has_builtin(__builtin_amdgcn_s_wait_tensorcnt)
#define USE_TDM 1
#endif

__device__ __forceinline__ unsigned short f2bf(float f) {
  unsigned u = __builtin_bit_cast(unsigned, f);
  unsigned r = u + 0x7FFFu + ((u >> 16) & 1u);      // round-to-nearest-even
  return (unsigned short)(r >> 16);
}

// Pack two fp32 -> packed 2 x bf16 (low = a, high = b), hw cvt_pk when available.
__device__ __forceinline__ unsigned pack2bf(float a, float b) {
#if __has_builtin(__builtin_amdgcn_cvt_pk_bf16_f32)
  typedef __attribute__((ext_vector_type(2))) __bf16 v2bf;
  v2bf p = __builtin_amdgcn_cvt_pk_bf16_f32(a, b);
  return __builtin_bit_cast(unsigned, p);
#else
  return (unsigned)f2bf(a) | ((unsigned)f2bf(b) << 16);
#endif
}

union FragBF {
  v16bf v;
  uint4 q[2];
};

// bf16 WMMA fragment from a K-major bf16 row (A: M=lane%16, B: N=lane%16).
// Elements 0..7 <- K=k0..k0+7 ; elements 8..15 <- K=k0+16..k0+23.
__device__ __forceinline__ void load_frag_bf16(FragBF& f, const unsigned short* __restrict__ row, int k0) {
  f.q[0] = *reinterpret_cast<const uint4*>(row + k0);
  f.q[1] = *reinterpret_cast<const uint4*>(row + k0 + 16);
}

__device__ __forceinline__ v8f wmma_bf16(const FragBF& a, const FragBF& b, v8f c) {
  return __builtin_amdgcn_wmma_f32_16x16x32_bf16(false, a.v, false, b.v, (short)0, c,
                                                 false, false);
}

__device__ __forceinline__ float sigmoid_f(float x) {
  x = fminf(fmaxf(x, -30.0f), 30.0f);
  return 1.0f / (1.0f + __expf(-x));
}
__device__ __forceinline__ float tanh_f(float x) {
  x = fminf(fmaxf(x, -15.0f), 15.0f);
  float e = __expf(-2.0f * x);
  return (1.0f - e) / (1.0f + e);
}

// Device-wide split barrier (persistent kernel, small resident grid).
__device__ __forceinline__ void grid_sync(unsigned* cnt, unsigned* gen, unsigned nb) {
  __threadfence();
  __syncthreads();
  if (threadIdx.x == 0) {
    unsigned g = __hip_atomic_load(gen, __ATOMIC_ACQUIRE, __HIP_MEMORY_SCOPE_AGENT);
    if (__hip_atomic_fetch_add(cnt, 1u, __ATOMIC_ACQ_REL, __HIP_MEMORY_SCOPE_AGENT) == nb - 1u) {
      __hip_atomic_store(cnt, 0u, __ATOMIC_RELAXED, __HIP_MEMORY_SCOPE_AGENT);
      __hip_atomic_fetch_add(gen, 1u, __ATOMIC_ACQ_REL, __HIP_MEMORY_SCOPE_AGENT);
    } else {
      while (__hip_atomic_load(gen, __ATOMIC_ACQUIRE, __HIP_MEMORY_SCOPE_AGENT) == g) {
        __builtin_amdgcn_s_sleep(2);
      }
    }
  }
  __syncthreads();
  __threadfence();
}

// ---------------------------------------------------------------------------
// Kernel 0: bf16 conversions (x gather, w_ih, w_hh) + zero h buffers/barrier.
// ---------------------------------------------------------------------------
__global__ __launch_bounds__(256) void init_kernel(
    const int* __restrict__ src, const float* __restrict__ emb,
    const float* __restrict__ w_ih, const float* __restrict__ w_hh,
    unsigned* __restrict__ x_bf, unsigned* __restrict__ wih_bf,
    unsigned* __restrict__ whh_bf, float* __restrict__ h_f32,
    unsigned* __restrict__ h_bf_u, unsigned* __restrict__ bar) {
  const long i = (long)blockIdx.x * blockDim.x + threadIdx.x;   // pair index

  const long NXP = (long)T_ * B_ * E_ / 2;                      // 4,194,304
  if (i < NXP) {
    const long m = i / (E_ / 2);
    const int  k2 = (int)(i % (E_ / 2));
    const int  tok = src[((int)m & 63) * T_ + ((int)m >> 6)];
    const float2 p = *reinterpret_cast<const float2*>(emb + (long)tok * E_ + k2 * 2);
    x_bf[i] = pack2bf(p.x, p.y);
  }
  if (i < (long)G3H * E_ / 2) {
    const float2 p = *reinterpret_cast<const float2*>(w_ih + i * 2);
    wih_bf[i] = pack2bf(p.x, p.y);
  }
  if (i < (long)G3H * H_ / 2) {
    const float2 p = *reinterpret_cast<const float2*>(w_hh + i * 2);
    whh_bf[i] = pack2bf(p.x, p.y);
  }
  if (i < B_ * H_) h_f32[i] = 0.0f;
  if (i < B_ * H_) h_bf_u[i] = 0u;    // 2 bufs as uints = 81,920
  if (i < 2)       bar[i] = 0u;
}

// ---------------------------------------------------------------------------
// Kernel 1: gi[t*B+b, :] = x_bf @ w_ih^T + b_ih  (all-bf16 operands)
// M = 8192, K = 1024, N = 3840. Wave tile 32x64 (2x4 WMMA accums),
// block = 8 waves (2M x 4N) = 64x256. Grid = 128 * 15 = 1920.
// ---------------------------------------------------------------------------
__global__ __launch_bounds__(256) void gi_gemm_kernel(
    const unsigned short* __restrict__ x_bf, const unsigned short* __restrict__ wih_bf,
    const float* __restrict__ b_ih, float* __restrict__ gi) {
  const int lane = threadIdx.x & 31;
  const int wave = threadIdx.x >> 5;
  const int lrow = lane & 15;
  const int lhi  = lane >> 4;

  const int bx = blockIdx.x & 127;            // M block (64 rows)
  const int by = blockIdx.x >> 7;             // N block (256 cols), 0..14
  const int Mbase = bx * 64 + (wave & 1) * 32;
  const int Nbase = by * 256 + (wave >> 1) * 64;

  const unsigned short* arow0 = x_bf + (long)(Mbase + lrow) * E_;
  const unsigned short* arow1 = arow0 + 16 * E_;
  const unsigned short* brow[4];
#pragma unroll
  for (int j = 0; j < 4; ++j)
    brow[j] = wih_bf + (long)(Nbase + j * 16 + lrow) * E_;

  const v8f vzero = {0.f, 0.f, 0.f, 0.f, 0.f, 0.f, 0.f, 0.f};
  v8f acc[2][4];
#pragma unroll
  for (int i = 0; i < 2; ++i)
#pragma unroll
    for (int j = 0; j < 4; ++j) acc[i][j] = vzero;

#pragma unroll 2
  for (int ck = 0; ck < E_ / 32; ++ck) {
    const int k0 = ck * 32 + lhi * 8;
    FragBF fa0, fa1, fb[4];
    load_frag_bf16(fa0, arow0, k0);
    load_frag_bf16(fa1, arow1, k0);
#pragma unroll
    for (int j = 0; j < 4; ++j) load_frag_bf16(fb[j], brow[j], k0);
#pragma unroll
    for (int j = 0; j < 4; ++j) {
      acc[0][j] = wmma_bf16(fa0, fb[j], acc[0][j]);
      acc[1][j] = wmma_bf16(fa1, fb[j], acc[1][j]);
    }
  }

  float bias[4];
#pragma unroll
  for (int j = 0; j < 4; ++j) bias[j] = b_ih[Nbase + j * 16 + lrow];
#pragma unroll
  for (int i = 0; i < 2; ++i) {
#pragma unroll
    for (int r = 0; r < 8; ++r) {
      const long row = (long)(Mbase + i * 16 + lhi * 8 + r) * G3H;
#pragma unroll
      for (int j = 0; j < 4; ++j)
        gi[row + Nbase + j * 16 + lrow] = acc[i][j][r] + bias[j];
    }
  }
}

// ---------------------------------------------------------------------------
// Kernel 2: persistent GRU scan. 20 blocks x 8 waves = 160 waves.
// Per step: TDM-stage the block's 32 h-rows (80 KB, contiguous) into LDS
// (fallback: cooperative b128 copy), then each wave runs 6 WMMA chains
// (2 M-tiles x 3 gates) with A from LDS, w_hh from L2; fp32 gates; one
// device-wide barrier per step.
// ---------------------------------------------------------------------------
__global__ __launch_bounds__(256) void gru_scan_kernel(
    const float* __restrict__ gi, const unsigned short* __restrict__ whh_bf,
    const float* __restrict__ b_hh, float* __restrict__ h_f32,
    unsigned short* __restrict__ h_bf, float* __restrict__ out,
    unsigned* __restrict__ bar) {
  extern __shared__ __align__(16) unsigned short shA[];   // 32 * H_ bf16 = 80 KB

  const int lane = threadIdx.x & 31;
  const int wave = threadIdx.x >> 5;
  const int lrow = lane & 15;
  const int lhi  = lane >> 4;
  const int ctile = (blockIdx.x % 10) * 8 + wave;   // 0..79
  const int Mbase = (blockIdx.x / 10) * 32;         // 0 or 32 (uniform per block)
  const int j = ctile * 16 + lrow;                  // this lane's h column

  const unsigned short* wr = whh_bf + (long)j * H_;
  const unsigned short* wz = wr + (long)H_ * H_;
  const unsigned short* wn = wz + (long)H_ * H_;
  const float bhr = b_hh[j], bhz = b_hh[H_ + j], bhn = b_hh[2 * H_ + j];
  const v8f vzero = {0.f, 0.f, 0.f, 0.f, 0.f, 0.f, 0.f, 0.f};

  const unsigned short* s0 = shA + lrow * H_;       // LDS A rows for M-tile 0
  const unsigned short* s1 = s0 + 16 * H_;          // and M-tile 1

  for (int t = 0; t < T_; ++t) {
    const int wbuf = t & 1;
    const unsigned short* hread = h_bf + (long)(1 - wbuf) * (B_ * H_);
    unsigned short* hwrite      = h_bf + (long)wbuf * (B_ * H_);

#ifdef USE_TDM
    // Tensor Data Mover: 1-D tile of 32*H_ = 40960 bf16 elements (80 KB)
    // global(hread + Mbase*H_) -> LDS offset 0 (dynamic-LDS base, no static LDS).
    if (wave == 0) {
      const unsigned long long ga =
          (unsigned long long)(const void*)(hread + (long)Mbase * H_);
      const u32x4 g0 = {1u,                               // count=1, is_restore=0
                        0u,                               // lds_addr = 0
                        (unsigned)ga,                     // global_addr[31:0]
                        (unsigned)((ga >> 32) & 0x01FFFFFFu) | 0x80000000u}; // addr[56:32] | type=2
      const i32x8 g1 = {0x00010000,        // workgroup_mask=0, data_size=1 (2 bytes)
                        (int)0xA0000000,   // tensor_dim0[15:0]=40960 in bits 63:48
                        0x00010000,        // tensor_dim0[31:16]=0, tensor_dim1=1
                        (int)0xA0000000,   // tile_dim0=40960 in bits 127:112
                        0,                 // tile_dim1=0 (1-D), tile_dim2=0
                        40960,             // tensor_dim0_stride[31:0]
                        0, 0};
      const i32x4 gz4 = {0, 0, 0, 0};
      const i32x8 gz8 = {0, 0, 0, 0, 0, 0, 0, 0};
      __builtin_amdgcn_tensor_load_to_lds(g0, g1, gz4, gz4, gz8, 0);
      __builtin_amdgcn_s_wait_tensorcnt(0);
    }
#else
    {
      const uint4* gsrc = reinterpret_cast<const uint4*>(hread + (long)Mbase * H_);
      uint4* ldst = reinterpret_cast<uint4*>(shA);
      for (int f = threadIdx.x; f < (32 * H_) / 8; f += 256) ldst[f] = gsrc[f];
    }
#endif
    __syncthreads();

    v8f aR[2], aZ[2], aN[2];
    aR[0] = vzero; aR[1] = vzero;
    aZ[0] = vzero; aZ[1] = vzero;
    aN[0] = vzero; aN[1] = vzero;

#pragma unroll 4
    for (int ck = 0; ck < H_ / 32; ++ck) {
      const int k0 = ck * 32 + lhi * 8;
      FragBF fa0, fa1, fr, fz, fn;
      load_frag_bf16(fa0, s0, k0);      // ds_load_b128 x2
      load_frag_bf16(fa1, s1, k0);
      load_frag_bf16(fr, wr, k0);       // global_load_b128 x2 (L2-resident)
      load_frag_bf16(fz, wz, k0);
      load_frag_bf16(fn, wn, k0);
      aR[0] = wmma_bf16(fa0, fr, aR[0]);
      aR[1] = wmma_bf16(fa1, fr, aR[1]);
      aZ[0] = wmma_bf16(fa0, fz, aZ[0]);
      aZ[1] = wmma_bf16(fa1, fz, aZ[1]);
      aN[0] = wmma_bf16(fa0, fn, aN[0]);
      aN[1] = wmma_bf16(fa1, fn, aN[1]);
    }

#pragma unroll
    for (int i = 0; i < 2; ++i) {
      const int rowb = Mbase + i * 16 + lhi * 8;
#pragma unroll
      for (int r = 0; r < 8; ++r) {
        const int b = rowb + r;
        const long gioff = ((long)t * B_ + b) * G3H;
        const float ir  = gi[gioff + j];
        const float iz  = gi[gioff + H_ + j];
        const float inn = gi[gioff + 2 * H_ + j];
        if (t + 1 < T_) {  // warm caches for next step's gi slice
          __builtin_prefetch(&gi[gioff + (long)B_ * G3H + j], 0, 1);
        }
        const float ghr = aR[i][r] + bhr;
        const float ghz = aZ[i][r] + bhz;
        const float ghn = aN[i][r] + bhn;
        const long hoff = (long)b * H_ + j;        // lane-private cell
        const float hold = h_f32[hoff];
        const float rg = sigmoid_f(ir + ghr);
        const float zg = sigmoid_f(iz + ghz);
        const float ng = tanh_f(inn + rg * ghn);
        const float hn = (1.0f - zg) * ng + zg * hold;
        h_f32[hoff] = hn;
        hwrite[hoff] = f2bf(hn);
        out[((long)t * B_ + b) * H_ + j] = hn;
      }
    }
    grid_sync(bar, bar + 1, NBLK2);
  }
}

// ---------------------------------------------------------------------------
extern "C" void kernel_launch(void* const* d_in, const int* in_sizes, int n_in,
                              void* d_out, int out_size, void* d_ws, size_t ws_size,
                              hipStream_t stream) {
  const int*   src  = (const int*)d_in[0];
  const float* emb  = (const float*)d_in[1];
  const float* w_ih = (const float*)d_in[2];
  const float* w_hh = (const float*)d_in[3];
  const float* b_ih = (const float*)d_in[4];
  const float* b_hh = (const float*)d_in[5];
  float* out = (float*)d_out;

  char* ws = (char*)d_ws;
  size_t off = 0;
  float* gi = (float*)(ws + off);                       off += (size_t)T_ * B_ * G3H * 4;  // 125.8 MB
  unsigned short* whh_bf = (unsigned short*)(ws + off); off += (size_t)G3H * H_ * 2;       // 9.8 MB
  unsigned short* wih_bf = (unsigned short*)(ws + off); off += (size_t)G3H * E_ * 2;       // 7.9 MB
  unsigned short* x_bf   = (unsigned short*)(ws + off); off += (size_t)T_ * B_ * E_ * 2;   // 16.8 MB
  float* h_f32 = (float*)(ws + off);                    off += (size_t)B_ * H_ * 4;
  unsigned short* h_bf = (unsigned short*)(ws + off);   off += (size_t)2 * B_ * H_ * 2;
  unsigned* bar = (unsigned*)(ws + off);                off += 256;

  hipLaunchKernelGGL(init_kernel, dim3(16384), dim3(256), 0, stream,
                     src, emb, w_ih, w_hh,
                     (unsigned*)x_bf, (unsigned*)wih_bf, (unsigned*)whh_bf,
                     h_f32, (unsigned*)h_bf, bar);
  hipLaunchKernelGGL(gi_gemm_kernel, dim3(1920), dim3(256), 0, stream,
                     x_bf, wih_bf, b_ih, gi);
  hipLaunchKernelGGL(gru_scan_kernel, dim3(NBLK2), dim3(256), 32 * H_ * 2, stream,
                     gi, whh_bf, b_hh, h_f32, h_bf, out, bar);
}
